// _AutoregressiveTransformerBlock_56745107915380
// MI455X (gfx1250) — matrix operand order repaired and checked by reading, not compiled, not run on hardware
//
#include <hip/hip_runtime.h>
#include <hip/hip_bf16.h>
#include <cstdint>

typedef __bf16 bf16;
typedef __attribute__((ext_vector_type(16))) __bf16 v16bf;
typedef __attribute__((ext_vector_type(8)))  float  v8f;

#define EMBED 1024
#define HEADS 16
#define HEAD_DIM 64
#define FF 4096
#define BB 8
#define TT 128
#define CARRY 4096
#define SEQ (CARRY + TT) /* 4224 */

enum { MODE_BF16 = 0, MODE_GELU = 1, MODE_Q = 2, MODE_KV = 3, MODE_RES = 4 };

static __device__ __forceinline__ v8f wmma_bf16(v16bf a, v16bf b, v8f c) {
  // D = A(16x32 bf16) x B(32x16 bf16) + C(16x16 f32) -> v_wmma_f32_16x16x32_bf16
  return __builtin_amdgcn_wmma_f32_16x16x32_bf16(false, a, false, b, (short)0, c,
                                                 false, false);
}

// ---------------------------------------------------------------------------
// LayerNorm: one block per row of [1024 x 1024]; writes bf16 for WMMA feeds.
// ---------------------------------------------------------------------------
__global__ __launch_bounds__(256) void ln_kernel(const float* __restrict__ x,
                                                 const float* __restrict__ sc,
                                                 const float* __restrict__ bi,
                                                 bf16* __restrict__ out) {
  const int row = blockIdx.x;
  const int t = threadIdx.x;
  const float4 v = reinterpret_cast<const float4*>(x + (size_t)row * EMBED)[t];
  float s = v.x + v.y + v.z + v.w;
  float q = v.x * v.x + v.y * v.y + v.z * v.z + v.w * v.w;
#pragma unroll
  for (int off = 16; off >= 1; off >>= 1) {
    s += __shfl_xor(s, off, 32);
    q += __shfl_xor(q, off, 32);
  }
  __shared__ float rS[8], rQ[8], mv[2];
  const int w = t >> 5, lane = t & 31;
  if (lane == 0) { rS[w] = s; rQ[w] = q; }
  __syncthreads();
  if (t == 0) {
    float S = 0.f, Q = 0.f;
    for (int i = 0; i < 8; ++i) { S += rS[i]; Q += rQ[i]; }
    const float mu = S * (1.0f / EMBED);
    const float var = Q * (1.0f / EMBED) - mu * mu;
    mv[0] = mu;
    mv[1] = rsqrtf(var + 1e-5f);
  }
  __syncthreads();
  const float mu = mv[0], inv = mv[1];
  const float4 scv = reinterpret_cast<const float4*>(sc)[t];
  const float4 biv = reinterpret_cast<const float4*>(bi)[t];
  bf16* o = out + (size_t)row * EMBED + t * 4;
  o[0] = (bf16)((v.x - mu) * inv * scv.x + biv.x);
  o[1] = (bf16)((v.y - mu) * inv * scv.y + biv.y);
  o[2] = (bf16)((v.z - mu) * inv * scv.z + biv.z);
  o[3] = (bf16)((v.w - mu) * inv * scv.w + biv.w);
}

// ---------------------------------------------------------------------------
// Carry -> k_all/v_all head copy ([B,H,4096,64] -> front of [B,H,4224,64]).
// Pure streaming float4; this is the HBM-bound part of the block.
// ---------------------------------------------------------------------------
__global__ __launch_bounds__(256) void copy_carry_kernel(
    const float4* __restrict__ ksrc, const float4* __restrict__ vsrc,
    float4* __restrict__ kdst, float4* __restrict__ vdst) {
  const size_t idx = (size_t)blockIdx.x * blockDim.x + threadIdx.x;  // < 128*65536
  const size_t bh = idx >> 16;      // / (4096*64/4)
  const size_t r  = idx & 65535;
  const size_t d  = bh * (size_t)(SEQ * HEAD_DIM / 4) + r;  // 67584 per (b,h)
  kdst[d] = ksrc[idx];
  vdst[d] = vsrc[idx];
}

// ---------------------------------------------------------------------------
// bf16 WMMA GEMM: out[M x N] = A[M x K](bf16) @ W[K x N](f32->bf16) + bias,
// fused epilogues. Block tile 128x64, 8 waves, each wave 16 rows x 64 cols.
// ---------------------------------------------------------------------------
__global__ __launch_bounds__(256) void gemm_kernel(
    const bf16* __restrict__ A, const float* __restrict__ W,
    const float* __restrict__ bias, int N, int K, int mode,
    bf16* __restrict__ outb, float* __restrict__ outf,
    const float* __restrict__ resin) {
  __shared__ __align__(16) bf16 Asm[128][32];  // A tile, row-major
  __shared__ __align__(16) bf16 Wsm[64][32];   // W tile, transposed [n][k]
  const int tid = threadIdx.x;
  const int w = tid >> 5, lane = tid & 31;
  const int nlane = lane & 15, hi = lane >> 4;
  const int N0 = blockIdx.x * 64;
  const int M0 = blockIdx.y * 128;

  v8f acc[4];
#pragma unroll
  for (int cc = 0; cc < 4; ++cc)
#pragma unroll
    for (int r = 0; r < 8; ++r) acc[cc][r] = 0.0f;

  for (int kb = 0; kb < K; kb += 32) {
    __syncthreads();
    {  // A tile: 128x32 bf16, 16 halves per thread
      const int row = tid >> 1, seg = (tid & 1) * 16;
      const uint4* src =
          reinterpret_cast<const uint4*>(A + (size_t)(M0 + row) * K + kb + seg);
      const uint4 u0 = src[0], u1 = src[1];
      *reinterpret_cast<uint4*>(&Asm[row][seg]) = u0;
      *reinterpret_cast<uint4*>(&Asm[row][seg + 8]) = u1;
    }
    {  // W tile: 32x64 f32 -> bf16 transposed, 8 floats per thread
      const int k = tid >> 3, n = (tid & 7) * 8;
      const float* wp = W + (size_t)(kb + k) * N + N0 + n;
      if (kb + 32 < K)
        __builtin_prefetch(wp + (size_t)32 * N, 0, 1);  // global_prefetch_b8
      const float4 f0 = reinterpret_cast<const float4*>(wp)[0];
      const float4 f1 = reinterpret_cast<const float4*>(wp)[1];
      Wsm[n + 0][k] = (bf16)f0.x; Wsm[n + 1][k] = (bf16)f0.y;
      Wsm[n + 2][k] = (bf16)f0.z; Wsm[n + 3][k] = (bf16)f0.w;
      Wsm[n + 4][k] = (bf16)f1.x; Wsm[n + 5][k] = (bf16)f1.y;
      Wsm[n + 6][k] = (bf16)f1.z; Wsm[n + 7][k] = (bf16)f1.w;
    }
    __syncthreads();

    // A operand: ISA 16-bit A 16x32 layout (m = lane&15, half = lane>>4)
    v16bf av;
    const int mrow = 16 * w + nlane;
#pragma unroll
    for (int e = 0; e < 16; ++e) {
      const int kk = (e & 7) + 16 * (e >> 3) + 8 * hi;
      av[e] = Asm[mrow][kk];
    }
#pragma unroll
    for (int cc = 0; cc < 4; ++cc) {
      v16bf bv;  // B operand: lane = column, K in table order (matches A)
#pragma unroll
      for (int e = 0; e < 16; ++e) bv[e] = Wsm[cc * 16 + nlane][(e & 7) + 16 * (e >> 3) + 8 * hi];
      acc[cc] = wmma_bf16(av, bv, acc[cc]);
    }
  }

  // Epilogue. D layout: lane n = lane&15, row m = r + 8*(lane>>4)
#pragma unroll
  for (int cc = 0; cc < 4; ++cc) {
#pragma unroll
    for (int r = 0; r < 8; ++r) {
      const int row = M0 + 16 * w + r + 8 * hi;
      const int col = N0 + cc * 16 + nlane;
      const float val = acc[cc][r] + bias[col];
      if (mode == MODE_BF16) {
        outb[(size_t)row * N + col] = (bf16)val;
      } else if (mode == MODE_GELU) {
        const float g = 0.5f * val * (1.0f + erff(val * 0.70710678118f));
        outb[(size_t)row * N + col] = (bf16)g;
      } else if (mode == MODE_Q) {
        const int b = row >> 7, tq = row & 127;
        const int h = col >> 6, d = col & 63;
        outb[(((size_t)b * HEADS + h) * TT + tq) * HEAD_DIM + d] = (bf16)val;
      } else if (mode == MODE_KV) {
        const int b = row >> 7, tq = row & 127;
        const int h = col >> 6, d = col & 63;
        outf[(((size_t)b * HEADS + h) * SEQ + CARRY + tq) * HEAD_DIM + d] = val;
      } else {  // MODE_RES (also used in-place for final FFN residual)
        const size_t i = (size_t)row * N + col;
        outf[i] = resin[i] + val;
      }
    }
  }
}

// ---------------------------------------------------------------------------
// Flash attention: block = (b,h); 8 waves = 8 q-tiles of 16 rows sharing
// 32-key K/V tiles in LDS (f32 -> bf16 on the fly, V transposed).
// Online softmax; ctx accumulated through v_wmma; output merged-head bf16.
// ---------------------------------------------------------------------------
__global__ __launch_bounds__(256) void attn_kernel(
    const bf16* __restrict__ qws, const float* __restrict__ kall,
    const float* __restrict__ vall, bf16* __restrict__ ctxws) {
  __shared__ __align__(16) bf16 Kt[32][64];       // [key][dim]
  __shared__ __align__(16) bf16 Vt[64][32];       // [dim][key] (transposed)
  __shared__ __align__(16) bf16 Pl[8][32][16];    // per-wave P: [key][row]
  const int tid = threadIdx.x;
  const int w = tid >> 5, lane = tid & 31;
  const int nlane = lane & 15, hi = lane >> 4;
  const int bh = blockIdx.x;
  const int b = bh >> 4, h = bh & 15;
  const int t0 = w * 16;

  // Q tile in registers, A-matrix layout (m = nlane, half = hi), 2 K-chunks
  const bf16* qrow = qws + ((size_t)bh * TT + t0 + nlane) * HEAD_DIM;
  v16bf qa[2];
#pragma unroll
  for (int c = 0; c < 2; ++c)
#pragma unroll
    for (int e = 0; e < 16; ++e) {
      const int d = c * 32 + (e & 7) + 16 * (e >> 3) + 8 * hi;
      qa[c][e] = qrow[d];
    }

  v8f ctxacc[4];
  float mrow[8], lsum[8];
#pragma unroll
  for (int cc = 0; cc < 4; ++cc)
#pragma unroll
    for (int r = 0; r < 8; ++r) ctxacc[cc][r] = 0.0f;
#pragma unroll
  for (int r = 0; r < 8; ++r) { mrow[r] = -3.0e38f; lsum[r] = 0.0f; }

  const float* kbase = kall + (size_t)bh * SEQ * HEAD_DIM;
  const float* vbase = vall + (size_t)bh * SEQ * HEAD_DIM;

  for (int s0 = 0; s0 < SEQ; s0 += 32) {
    __syncthreads();
#pragma unroll
    for (int u = 0; u < 2; ++u) {  // 512 float4 per tensor / 256 threads
      const int idx4 = tid + u * 256;
      const int sr = idx4 >> 4;          // key row 0..31
      const int j = (idx4 & 15) * 4;     // dim 0..60
      const float4 kv = reinterpret_cast<const float4*>(
          kbase + (size_t)(s0 + sr) * HEAD_DIM)[idx4 & 15];
      Kt[sr][j + 0] = (bf16)kv.x; Kt[sr][j + 1] = (bf16)kv.y;
      Kt[sr][j + 2] = (bf16)kv.z; Kt[sr][j + 3] = (bf16)kv.w;
      const float4 vv = reinterpret_cast<const float4*>(
          vbase + (size_t)(s0 + sr) * HEAD_DIM)[idx4 & 15];
      Vt[j + 0][sr] = (bf16)vv.x; Vt[j + 1][sr] = (bf16)vv.y;
      Vt[j + 2][sr] = (bf16)vv.z; Vt[j + 3][sr] = (bf16)vv.w;
    }
    __syncthreads();

    // scores for two 16-key tiles: S = Q @ K^T
    float sv[2][8];
#pragma unroll
    for (int nt = 0; nt < 2; ++nt) {
      v8f sacc;
#pragma unroll
      for (int r = 0; r < 8; ++r) sacc[r] = 0.0f;
#pragma unroll
      for (int c = 0; c < 2; ++c) {
        v16bf kbv;  // B[d][n] = Kt[key n][d] -> d in table order (matches A)
#pragma unroll
        for (int e = 0; e < 16; ++e)
          kbv[e] = Kt[nt * 16 + nlane][c * 32 + (e & 7) + 16 * (e >> 3) + 8 * hi];
        sacc = wmma_bf16(qa[c], kbv, sacc);
      }
      const int kg = s0 + nt * 16 + nlane;
#pragma unroll
      for (int r = 0; r < 8; ++r) {
        const float scf = sacc[r] * 0.125f;  // 1/sqrt(64)
        const int qpos = CARRY + t0 + r + 8 * hi;
        sv[nt][r] = (kg <= qpos) ? scf : -1.0e30f;
      }
    }

    // online softmax per row (row-group = 16 lanes with same hi)
#pragma unroll
    for (int r = 0; r < 8; ++r) {
      float vm = fmaxf(sv[0][r], sv[1][r]);
#pragma unroll
      for (int off = 8; off >= 1; off >>= 1) vm = fmaxf(vm, __shfl_xor(vm, off, 32));
      const float mnew = fmaxf(mrow[r], vm);
      const float corr = __expf(mrow[r] - mnew);
      const float p0 = __expf(sv[0][r] - mnew);
      const float p1 = __expf(sv[1][r] - mnew);
      float rsum = p0 + p1;
#pragma unroll
      for (int off = 8; off >= 1; off >>= 1) rsum += __shfl_xor(rsum, off, 32);
      lsum[r] = lsum[r] * corr + rsum;
      mrow[r] = mnew;
#pragma unroll
      for (int cc = 0; cc < 4; ++cc) ctxacc[cc][r] *= corr;
      const int m = r + 8 * hi;
      Pl[w][nlane][m] = (bf16)p0;        // D-layout -> [key][row] staging
      Pl[w][16 + nlane][m] = (bf16)p1;
    }

    // P as A operand (16 rows x 32 keys), then ctx += P @ V
    v16bf pa;
#pragma unroll
    for (int e = 0; e < 16; ++e) {
      const int kk = (e & 7) + 16 * (e >> 3) + 8 * hi;
      pa[e] = Pl[w][kk][nlane];
    }
#pragma unroll
    for (int cc = 0; cc < 4; ++cc) {
      v16bf vb;  // B[k][n] = Vt[dim n][key k] -> key in table order (matches A)
#pragma unroll
      for (int e = 0; e < 16; ++e) vb[e] = Vt[cc * 16 + nlane][(e & 7) + 16 * (e >> 3) + 8 * hi];
      ctxacc[cc] = wmma_bf16(pa, vb, ctxacc[cc]);
    }
  }

  // normalize + merged-head store: ctxws[(b*T+t)*E + h*64 + d]
#pragma unroll
  for (int cc = 0; cc < 4; ++cc)
#pragma unroll
    for (int r = 0; r < 8; ++r) {
      const int m = r + 8 * hi;
      const float o = ctxacc[cc][r] / lsum[r];
      ctxws[((size_t)b * TT + t0 + m) * EMBED + h * HEAD_DIM + cc * 16 + nlane] =
          (bf16)o;
    }
}

// ---------------------------------------------------------------------------
extern "C" void kernel_launch(void* const* d_in, const int* in_sizes, int n_in,
                              void* d_out, int out_size, void* d_ws,
                              size_t ws_size, hipStream_t stream) {
  (void)in_sizes; (void)n_in; (void)out_size; (void)ws_size;
  const float* hidden      = (const float*)d_in[0];
  const float* key_carry   = (const float*)d_in[1];
  const float* value_carry = (const float*)d_in[2];
  const float* ln1_s = (const float*)d_in[3];
  const float* ln1_b = (const float*)d_in[4];
  const float* wq = (const float*)d_in[5];
  const float* bq = (const float*)d_in[6];
  const float* wk = (const float*)d_in[7];
  const float* bk = (const float*)d_in[8];
  const float* wv = (const float*)d_in[9];
  const float* bv = (const float*)d_in[10];
  const float* wo = (const float*)d_in[11];
  const float* bo = (const float*)d_in[12];
  const float* ln2_s = (const float*)d_in[13];
  const float* ln2_b = (const float*)d_in[14];
  const float* w1 = (const float*)d_in[15];
  const float* b1 = (const float*)d_in[16];
  const float* w2 = (const float*)d_in[17];
  const float* b2 = (const float*)d_in[18];

  float* out_hidden = (float*)d_out;                                   // [8,128,1024]
  float* k_all = out_hidden + (size_t)BB * TT * EMBED;                 // [8,16,4224,64]
  float* v_all = k_all + (size_t)BB * HEADS * SEQ * HEAD_DIM;

  char* ws = (char*)d_ws;
  bf16* x_bf   = (bf16*)(ws);                    // 2 MB  LN1 out (bf16)
  bf16* q_ws   = (bf16*)(ws + (2u << 20));       // 2 MB  Q [B,H,T,64]
  bf16* ctx_ws = (bf16*)(ws + (4u << 20));       // 2 MB  ctx merged heads
  bf16* f_bf   = (bf16*)(ws + (6u << 20));       // 2 MB  LN2 out
  bf16* ffn_ws = (bf16*)(ws + (8u << 20));       // 8 MB  GELU(f@w1+b1)

  const int M = BB * TT;  // 1024
  const dim3 blk(256);
  const dim3 gE(EMBED / 64, M / 128);  // N=1024 GEMMs
  const dim3 gF(FF / 64, M / 128);     // N=4096 GEMM

  ln_kernel<<<M, blk, 0, stream>>>(hidden, ln1_s, ln1_b, x_bf);

  gemm_kernel<<<gE, blk, 0, stream>>>(x_bf, wq, bq, EMBED, EMBED, MODE_Q,
                                      q_ws, nullptr, nullptr);
  gemm_kernel<<<gE, blk, 0, stream>>>(x_bf, wk, bk, EMBED, EMBED, MODE_KV,
                                      nullptr, k_all, nullptr);
  gemm_kernel<<<gE, blk, 0, stream>>>(x_bf, wv, bv, EMBED, EMBED, MODE_KV,
                                      nullptr, v_all, nullptr);

  copy_carry_kernel<<<32768, blk, 0, stream>>>(
      (const float4*)key_carry, (const float4*)value_carry,
      (float4*)k_all, (float4*)v_all);

  attn_kernel<<<BB * HEADS, blk, 0, stream>>>(q_ws, k_all, v_all, ctx_ws);

  gemm_kernel<<<gE, blk, 0, stream>>>(ctx_ws, wo, bo, EMBED, EMBED, MODE_RES,
                                      nullptr, out_hidden, hidden);

  ln_kernel<<<M, blk, 0, stream>>>(out_hidden, ln2_s, ln2_b, f_bf);

  gemm_kernel<<<gF, blk, 0, stream>>>(f_bf, w1, b1, FF, EMBED, MODE_GELU,
                                      ffn_ws, nullptr, nullptr);
  gemm_kernel<<<gE, blk, 0, stream>>>(ffn_ws, w2, b2, EMBED, FF, MODE_RES,
                                      nullptr, out_hidden, out_hidden);
}
